// PerceiverAttention_18829136626080
// MI455X (gfx1250) — matrix operand order repaired and checked by reading, not compiled, not run on hardware
//
#include <hip/hip_runtime.h>

typedef __attribute__((ext_vector_type(16))) __bf16 v16bf;
typedef __attribute__((ext_vector_type(8)))  float  v8f;
typedef __attribute__((ext_vector_type(4)))  unsigned u32x4;
typedef __attribute__((ext_vector_type(4)))  int      i32x4;
typedef __attribute__((ext_vector_type(8)))  int      i32x8;

union BFrag { unsigned u[8]; v16bf v; };
union FAcc  { float f[8];    v8f   v; };

#define DIMD   1024
#define HEADS  16
#define DHEAD  64
#define N1     1024
#define N2     64
#define NKV    1088     // N1 + N2
#define BTC    32       // b * T
#define QSCALE 0.125f   // 64^-0.5

// ---------- bf16 helpers (bf16 kept as raw u16; no scalar __bf16 math) ----------
__device__ __forceinline__ unsigned short f2bf(float f) {
  unsigned u = __float_as_uint(f);
  u += 0x7FFFu + ((u >> 16) & 1u);          // round-to-nearest-even
  return (unsigned short)(u >> 16);
}

__device__ __forceinline__ v8f wmma_bf16(v16bf a, v16bf b, v8f c) {
  return __builtin_amdgcn_wmma_f32_16x16x32_bf16(false, a, false, b, (short)0, c, false, false);
}

// A fragment: 16(M)x32(K) bf16 tile at p, row-major with `stride` elems/row.
// ISA layout: lanes 0-15 -> M=lane, K pairs 0..7 & 16..23; lanes 16-31 -> K+8.
__device__ __forceinline__ v16bf load_a16(const unsigned short* p, int stride, int lane) {
  const unsigned short* r = p + (lane & 15) * stride + ((lane & 16) ? 8 : 0);
  BFrag f;
#pragma unroll
  for (int v = 0; v < 4; ++v) {
    f.u[v]     = *(const unsigned*)(r + 2 * v);
    f.u[v + 4] = *(const unsigned*)(r + 16 + 2 * v);
  }
  return f.v;
}

// B fragment: 32(K)x16(N) bf16 tile at p, row-major with `stride` elems/row.
// ISA layout: lanes 0-15 -> N=lane, K pairs 0..7 & 16..23; lanes 16-31 -> K+8.
__device__ __forceinline__ v16bf load_b16(const unsigned short* p, int stride, int lane) {
  const unsigned short* r = p + (lane & 15) + ((lane & 16) ? 8 : 0) * stride;
  BFrag f;
#pragma unroll
  for (int v = 0; v < 4; ++v) {
    f.u[v]     = (unsigned)r[(2 * v) * stride] | ((unsigned)r[(2 * v + 1) * stride] << 16);
    f.u[v + 4] = (unsigned)r[(16 + 2 * v) * stride] | ((unsigned)r[(17 + 2 * v) * stride] << 16);
  }
  return f.v;
}

// ---------- Tensor Data Mover: 2D bf16 tile -> LDS (D# per CDNA5 ISA) ----------
// tileW/tileH in 2-byte elements; strideW = global row stride in elements.
// Tensor dims set equal to tile dims (tiles always fully in-bounds here).
__device__ __forceinline__ void tdm_load_2d(unsigned ldsAddr, const void* gptr,
                                            unsigned tileW, unsigned tileH,
                                            unsigned long long strideW,
                                            bool padEnable, unsigned padIntervalCode,
                                            unsigned padAmountCode) {
  const unsigned long long ga = (unsigned long long)(size_t)gptr;
  u32x4 g0;
  g0[0] = 1u;                                               // count=1, user descriptor
  g0[1] = ldsAddr;                                          // lds_addr (bytes)
  g0[2] = (unsigned)(ga & 0xFFFFFFFFull);                   // global_addr[31:0]
  g0[3] = (unsigned)((ga >> 32) & 0x01FFFFFFull) | (2u << 30);  // global_addr[56:32] | type=2

  unsigned d0 = (1u << 16);                                 // data_size=1 (2B), wg_mask=0
  if (padEnable) d0 |= (1u << 20) | (padIntervalCode << 22) | (padAmountCode << 25);
  const unsigned long long s0 = strideW;                    // tensor_dim0_stride (elements)
  i32x8 g1;
  g1[0] = (int)d0;
  g1[1] = (int)((tileW & 0xFFFFu) << 16);                   // tensor_dim0[15:0] @ [63:48]
  g1[2] = (int)(((tileW >> 16) & 0xFFFFu) | ((tileH & 0xFFFFu) << 16));   // td0 hi | td1 lo
  g1[3] = (int)(((tileH >> 16) & 0xFFFFu) | ((tileW & 0xFFFFu) << 16));   // td1 hi | tile_dim0
  g1[4] = (int)(tileH & 0xFFFFu);                           // tile_dim1 (tile_dim2 = 0)
  g1[5] = (int)(s0 & 0xFFFFFFFFull);                        // stride0[31:0]
  g1[6] = (int)((s0 >> 32) & 0xFFFFull);                    // stride0[47:32] (stride1 = 0)
  g1[7] = 0;

  const i32x4 z4 = {0, 0, 0, 0};
#if __clang_major__ >= 23
  const i32x8 z8 = {0, 0, 0, 0, 0, 0, 0, 0};
  __builtin_amdgcn_tensor_load_to_lds(g0, g1, z4, z4, z8, 0);
#else
  __builtin_amdgcn_tensor_load_to_lds(g0, g1, z4, z4, 0);
#endif
}

// ---------- fp32 -> bf16 weight conversion ----------
__global__ __launch_bounds__(256) void cvt_f32_bf16(const float* __restrict__ in,
                                                    unsigned short* __restrict__ out, int n) {
  int i = blockIdx.x * 256 + threadIdx.x;
  if (i < n) out[i] = f2bf(in[i]);
}

// ---------- fused LayerNorm + bf16 cast; one row (1024) per 256-thread block ----------
__global__ __launch_bounds__(256) void ln_bf16(const float* __restrict__ X,
                                               const float* __restrict__ gamma,
                                               const float* __restrict__ beta,
                                               unsigned short* __restrict__ dst,
                                               unsigned short* __restrict__ dst2,
                                               int rowsPerBT, int rowOffset) {
  const int r  = blockIdx.x;
  const int bt = r / rowsPerBT;
  const int i  = r - bt * rowsPerBT;
  const int tid = threadIdx.x;

  const float4 xv = ((const float4*)(X + (size_t)r * DIMD))[tid];
  float s  = xv.x + xv.y + xv.z + xv.w;
  float ss = xv.x * xv.x + xv.y * xv.y + xv.z * xv.z + xv.w * xv.w;
#pragma unroll
  for (int m = 1; m < 32; m <<= 1) {
    s  += __shfl_xor(s, m, 32);
    ss += __shfl_xor(ss, m, 32);
  }
  __shared__ float rs[8], rss[8];
  if ((tid & 31) == 0) { rs[tid >> 5] = s; rss[tid >> 5] = ss; }
  __syncthreads();
  float S = 0.f, SS = 0.f;
#pragma unroll
  for (int w = 0; w < 8; ++w) { S += rs[w]; SS += rss[w]; }
  const float mu   = S * (1.0f / DIMD);
  const float var  = SS * (1.0f / DIMD) - mu * mu;
  const float rstd = rsqrtf(var + 1e-5f);

  const float4 gv = ((const float4*)gamma)[tid];
  const float4 bv = ((const float4*)beta)[tid];
  unsigned short o[4];
  o[0] = f2bf((xv.x - mu) * rstd * gv.x + bv.x);
  o[1] = f2bf((xv.y - mu) * rstd * gv.y + bv.y);
  o[2] = f2bf((xv.z - mu) * rstd * gv.z + bv.z);
  o[3] = f2bf((xv.w - mu) * rstd * gv.w + bv.w);

  unsigned short* d = dst + ((size_t)(bt * NKV + rowOffset + i)) * DIMD + tid * 4;
#pragma unroll
  for (int k = 0; k < 4; ++k) d[k] = o[k];
  if (dst2) {
    unsigned short* d2 = dst2 + (size_t)r * DIMD + tid * 4;
#pragma unroll
    for (int k = 0; k < 4; ++k) d2[k] = o[k];
  }
}

// ---------- bf16 WMMA GEMM with TDM double-buffered staging ----------
// C[M,N] = alpha * A[M,K] @ B[K,N]; 8 waves; workgroup tile 128x128; wave 32x64; K step 32.
// Wave 0 drives the Tensor Data Mover: 2 tensor_load_to_lds per stage, 2 stages in flight.
// Requires M%128==0, N%128==0, K%32==0 (true for every call here).
template <bool OUT_BF16>
__global__ __launch_bounds__(256) void gemm_bf16(const unsigned short* __restrict__ A,
                                                 const unsigned short* __restrict__ B,
                                                 void* __restrict__ C,
                                                 int M, int N, int K, float alpha) {
  __shared__ unsigned short sA[2][128][32];        // 2 x 8 KB (TDM fills, no pad)
  __shared__ unsigned short sB[2][32][128 + 8];    // 2 x 8.5 KB (TDM pad: 64 dw -> +4 dw)

  const int tid  = threadIdx.x;
  const int lane = tid & 31;
  const int wave = tid >> 5;        // 0..7
  const int wm   = wave >> 1;       // 0..3 : M offset 32*wm
  const int wn   = wave & 1;        // 0..1 : N offset 64*wn
  const int blockM = blockIdx.y * 128;
  const int blockN = blockIdx.x * 128;

  FAcc acc[2][4];
#pragma unroll
  for (int mf = 0; mf < 2; ++mf)
#pragma unroll
    for (int nf = 0; nf < 4; ++nf)
#pragma unroll
      for (int v = 0; v < 8; ++v) acc[mf][nf].f[v] = 0.f;

  const int nIter = K / 32;
  auto issue = [&](int it) {
    const int buf = it & 1;
    const int k0  = it * 32;
    // A tile: 128 rows x 32 K-elems, global row stride K
    tdm_load_2d((unsigned)(size_t)&sA[buf][0][0],
                A + (size_t)blockM * K + k0,
                32u, 128u, (unsigned long long)K,
                false, 0u, 0u);
    // B tile: 32 rows x 128 N-elems, global row stride N; LDS pad 4 dwords per 64 dwords
    tdm_load_2d((unsigned)(size_t)&sB[buf][0][0],
                B + (size_t)k0 * N + blockN,
                128u, 32u, (unsigned long long)N,
                true, 5u /*64 dwords*/, 3u /*4 dwords*/);
  };

  if (wave == 0) issue(0);

  for (int it = 0; it < nIter; ++it) {
    const int buf = it & 1;
    if (wave == 0) {
      if (it + 1 < nIter) {
        issue(it + 1);
        __builtin_amdgcn_s_wait_tensorcnt(2);   // current stage's 2 DMAs complete
      } else {
        __builtin_amdgcn_s_wait_tensorcnt(0);
      }
    }
    __syncthreads();

    const v16bf a0 = load_a16(&sA[buf][wm * 32][0],      32, lane);
    const v16bf a1 = load_a16(&sA[buf][wm * 32 + 16][0], 32, lane);
#pragma unroll
    for (int nf = 0; nf < 4; ++nf) {
      const v16bf b = load_b16(&sB[buf][0][wn * 64 + nf * 16], 128 + 8, lane);
      acc[0][nf].v = wmma_bf16(a0, b, acc[0][nf].v);
      acc[1][nf].v = wmma_bf16(a1, b, acc[1][nf].v);
    }
    __syncthreads();
  }

  // C/D layout: VGPR v -> M row, lane -> N col (lanes>=16 shift M by 8)
  const int nl = lane & 15;
  const int mo = (lane >> 4) * 8;
#pragma unroll
  for (int mf = 0; mf < 2; ++mf)
#pragma unroll
    for (int nf = 0; nf < 4; ++nf)
#pragma unroll
      for (int v = 0; v < 8; ++v) {
        const size_t row = (size_t)(blockM + wm * 32 + mf * 16 + mo + v);
        const size_t col = (size_t)(blockN + wn * 64 + nf * 16 + nl);
        const float val = acc[mf][nf].f[v] * alpha;
        if (OUT_BF16) ((unsigned short*)C)[row * N + col] = f2bf(val);
        else          ((float*)C)[row * N + col]          = val;
      }
}

// ---------- flash attention: one block per (b*t, head); 4 waves x 16 query rows ----------
__global__ __launch_bounds__(128) void flash_attn(const unsigned short* __restrict__ Q,
                                                  const unsigned short* __restrict__ KV,
                                                  unsigned short* __restrict__ O) {
  const int bt   = blockIdx.x;   // 0..31
  const int h    = blockIdx.y;   // 0..15
  const int tid  = threadIdx.x;
  const int lane = tid & 31;
  const int warp = tid >> 5;     // 0..3 -> query rows warp*16..+15

  __shared__ unsigned short sKt[64][64 + 8];   // K block, transposed: [d][j]
  __shared__ unsigned short sV [64][64 + 8];   // V block: [j][d]
  __shared__ unsigned short sP [64][64 + 8];   // P = exp(S-m): [m][j]

  // ---- Q strip as two A fragments (head-dim K=64 -> 2 steps of 32); SCALE pre-folded in q-proj
  const unsigned short* qbase = Q + (size_t)(bt * N2) * DIMD + h * DHEAD;
  v16bf aQ[2];
  {
    const unsigned short* r = qbase + (size_t)(warp * 16 + (lane & 15)) * DIMD + ((lane & 16) ? 8 : 0);
#pragma unroll
    for (int kd = 0; kd < 2; ++kd) {
      BFrag f;
#pragma unroll
      for (int v = 0; v < 4; ++v) {
        f.u[v]     = *(const unsigned*)(r + kd * 32 + 2 * v);
        f.u[v + 4] = *(const unsigned*)(r + kd * 32 + 16 + 2 * v);
      }
      aQ[kd] = f.v;
    }
  }

  float m_run[8], l_run[8];
  FAcc o[4];
#pragma unroll
  for (int v = 0; v < 8; ++v) { m_run[v] = -3.0e38f; l_run[v] = 0.f; }
#pragma unroll
  for (int nf = 0; nf < 4; ++nf)
#pragma unroll
    for (int v = 0; v < 8; ++v) o[nf].f[v] = 0.f;

  const unsigned short* kvbase = KV + (size_t)(bt * NKV) * (2 * DIMD) + h * DHEAD;

  for (int jb = 0; jb < NKV; jb += 64) {
    __syncthreads();                          // previous PV reads done
    {   // stage K (transposed) and V blocks: thread -> (j = tid/2, d half)
      const int j  = tid >> 1;
      const int d0 = (tid & 1) * 32;
      const unsigned short* kr = kvbase + (size_t)(jb + j) * (2 * DIMD) + d0;
#pragma unroll
      for (int d = 0; d < 32; ++d) sKt[d0 + d][j] = kr[d];
      const unsigned short* vr = kr + DIMD;   // v = kv[..., 1024 + h*64 + d]
#pragma unroll
      for (int q = 0; q < 4; ++q)
        *(uint4*)&sV[j][d0 + 8 * q] = *(const uint4*)(vr + 8 * q);
    }
    __syncthreads();

    if (jb + 64 < NKV)                        // prefetch next KV block
      __builtin_prefetch((const void*)(kvbase + (size_t)(jb + 64 + (tid & 63)) * (2 * DIMD)), 0, 1);

    // ---- S strip = Q_strip (16x64) @ K_block^T  -> 16x64 in 4 N-tiles
    FAcc s[4];
#pragma unroll
    for (int nf = 0; nf < 4; ++nf)
#pragma unroll
      for (int v = 0; v < 8; ++v) s[nf].f[v] = 0.f;
#pragma unroll
    for (int kd = 0; kd < 2; ++kd)
#pragma unroll
      for (int nf = 0; nf < 4; ++nf) {
        const v16bf bK = load_b16(&sKt[kd * 32][nf * 16], 64 + 8, lane);
        s[nf].v = wmma_bf16(aQ[kd], bK, s[nf].v);
      }

    // ---- online softmax (rows live as: frag elem v -> row mo+v, 16-lane N groups)
    float mb[8];
#pragma unroll
    for (int v = 0; v < 8; ++v) {
      mb[v] = s[0].f[v];
#pragma unroll
      for (int nf = 1; nf < 4; ++nf) mb[v] = fmaxf(mb[v], s[nf].f[v]);
    }
#pragma unroll
    for (int m = 1; m < 16; m <<= 1)
#pragma unroll
      for (int v = 0; v < 8; ++v) mb[v] = fmaxf(mb[v], __shfl_xor(mb[v], m, 32));

    float mn[8], sc[8], ps[8];
#pragma unroll
    for (int v = 0; v < 8; ++v) {
      mn[v] = fmaxf(m_run[v], mb[v]);
      sc[v] = __expf(m_run[v] - mn[v]);
      ps[v] = 0.f;
    }
    const int nl = lane & 15;
    const int mo = (lane >> 4) * 8;
#pragma unroll
    for (int nf = 0; nf < 4; ++nf)
#pragma unroll
      for (int v = 0; v < 8; ++v) {
        const float p = __expf(s[nf].f[v] - mn[v]);
        ps[v] += p;
        sP[warp * 16 + mo + v][nf * 16 + nl] = f2bf(p);
      }
#pragma unroll
    for (int m = 1; m < 16; m <<= 1)
#pragma unroll
      for (int v = 0; v < 8; ++v) ps[v] += __shfl_xor(ps[v], m, 32);
#pragma unroll
    for (int v = 0; v < 8; ++v) {
      l_run[v] = l_run[v] * sc[v] + ps[v];
      m_run[v] = mn[v];
    }
#pragma unroll
    for (int nf = 0; nf < 4; ++nf)
#pragma unroll
      for (int v = 0; v < 8; ++v) o[nf].f[v] *= sc[v];

    __syncthreads();                          // sP visible (layout shuffle via LDS)

    // ---- O += P (16x64) @ V_block (64x64)
#pragma unroll
    for (int kd = 0; kd < 2; ++kd) {
      const v16bf aP = load_a16(&sP[warp * 16][kd * 32], 64 + 8, lane);
#pragma unroll
      for (int nf = 0; nf < 4; ++nf) {
        const v16bf bV = load_b16(&sV[kd * 32][nf * 16], 64 + 8, lane);
        o[nf].v = wmma_bf16(aP, bV, o[nf].v);
      }
    }
  }

  // ---- normalize and write bf16 (b t n (h d)) for the final Wo GEMM
  unsigned short* obase = O + (size_t)(bt * N2) * DIMD + h * DHEAD;
  const int nl = lane & 15;
  const int mo = (lane >> 4) * 8;
#pragma unroll
  for (int nf = 0; nf < 4; ++nf)
#pragma unroll
    for (int v = 0; v < 8; ++v) {
      const int m = warp * 16 + mo + v;
      const int d = nf * 16 + nl;
      obase[(size_t)m * DIMD + d] = f2bf(o[nf].f[v] / l_run[v]);
    }
}

// ---------------------------------------------------------------------------
extern "C" void kernel_launch(void* const* d_in, const int* in_sizes, int n_in,
                              void* d_out, int out_size, void* d_ws, size_t ws_size,
                              hipStream_t stream) {
  const float* x   = (const float*)d_in[0];
  const float* lat = (const float*)d_in[1];
  const float* gm  = (const float*)d_in[2];
  const float* bm  = (const float*)d_in[3];
  const float* gl  = (const float*)d_in[4];
  const float* bl  = (const float*)d_in[5];
  const float* Wq  = (const float*)d_in[6];
  const float* Wkv = (const float*)d_in[7];
  const float* Wo  = (const float*)d_in[8];
  float* out = (float*)d_out;

  char* ws = (char*)d_ws;
  size_t off = 0;
  auto carve = [&](size_t bytes) {
    char* p = ws + off;
    off += (bytes + 255) & ~(size_t)255;
    return p;
  };
  unsigned short* concatB = (unsigned short*)carve((size_t)BTC * NKV * DIMD * 2);        // LN'd [x; lat]
  unsigned short* qin     = (unsigned short*)carve((size_t)BTC * N2  * DIMD * 2);        // LN'd latents
  unsigned short* WqB     = (unsigned short*)carve((size_t)DIMD * DIMD * 2);
  unsigned short* WkvB    = (unsigned short*)carve((size_t)DIMD * 2 * DIMD * 2);
  unsigned short* WoB     = (unsigned short*)carve((size_t)DIMD * DIMD * 2);
  unsigned short* qB      = (unsigned short*)carve((size_t)BTC * N2  * DIMD * 2);        // q (scaled)
  unsigned short* kvB     = (unsigned short*)carve((size_t)BTC * NKV * 2 * DIMD * 2);    // [k|v]
  unsigned short* attnB   = (unsigned short*)carve((size_t)BTC * N2  * DIMD * 2);        // attn out

  // 1) weights -> bf16
  cvt_f32_bf16<<<(DIMD * DIMD + 255) / 256, 256, 0, stream>>>(Wq, WqB, DIMD * DIMD);
  cvt_f32_bf16<<<(DIMD * 2 * DIMD + 255) / 256, 256, 0, stream>>>(Wkv, WkvB, DIMD * 2 * DIMD);
  cvt_f32_bf16<<<(DIMD * DIMD + 255) / 256, 256, 0, stream>>>(Wo, WoB, DIMD * DIMD);

  // 2) fused LayerNorm -> bf16
  ln_bf16<<<BTC * N1, 256, 0, stream>>>(x,   gm, bm, concatB, nullptr, N1, 0);
  ln_bf16<<<BTC * N2, 256, 0, stream>>>(lat, gl, bl, concatB, qin,     N2, N1);

  // 3) q = latn @ Wq * SCALE   (M=2048, N=1024, K=1024)
  gemm_bf16<true><<<dim3(DIMD / 128, (BTC * N2) / 128), 256, 0, stream>>>(
      qin, WqB, qB, BTC * N2, DIMD, DIMD, QSCALE);

  // 4) kv = [xn; latn] @ Wkv   (M=34816, N=2048, K=1024)
  gemm_bf16<true><<<dim3((2 * DIMD) / 128, (BTC * NKV) / 128), 256, 0, stream>>>(
      concatB, WkvB, kvB, BTC * NKV, 2 * DIMD, DIMD, 1.0f);

  // 5) flash attention per (b*t, head)
  flash_attn<<<dim3(BTC, HEADS), 128, 0, stream>>>(qB, kvB, attnB);

  // 6) out = attn @ Wo -> fp32 d_out   (M=2048, N=1024, K=1024)
  gemm_bf16<false><<<dim3(DIMD / 128, (BTC * N2) / 128), 256, 0, stream>>>(
      attnB, WoB, out, BTC * N2, DIMD, DIMD, 1.0f);

  (void)in_sizes; (void)n_in; (void)out_size; (void)ws_size;
}
